// SparsemaxGenerator_14508399526052
// MI455X (gfx1250) — compile-verified, matrix operand order checked by reading, not run on hardware
//
#include <hip/hip_runtime.h>

// ---------------------------------------------------------------------------
// Problem constants (match reference)
// ---------------------------------------------------------------------------
#define BB   16
#define SS   2048
#define EE   512
#define HH   512
#define ENC  1024
#define N4H  2048   // 4*H gate width

typedef __attribute__((ext_vector_type(16))) __bf16 v16bf;
typedef __attribute__((ext_vector_type(8)))  float  v8f;

// ---------------------------------------------------------------------------
// CDNA5 async global->LDS copy (ASYNCcnt path) + wait
// LDS byte address = low 32 bits of the flat address (LDS aperture keeps the
// offset in addr[31:0], ISA 10.2).
// ---------------------------------------------------------------------------
__device__ __forceinline__ unsigned lds_addr(const void* p) {
  return (unsigned)(size_t)p;
}
__device__ __forceinline__ void async_g2l_b128(unsigned lds_byte, const void* gptr) {
  asm volatile("global_load_async_to_lds_b128 %0, %1, off"
               :: "v"(lds_byte), "v"(gptr) : "memory");
}
__device__ __forceinline__ void wait_async0() {
  asm volatile("s_wait_asynccnt 0" ::: "memory");
}

// ---------------------------------------------------------------------------
// Fragment loaders: lane<16 -> row=lane, cols kk..kk+15 ; lane>=16 ->
// row=lane-16, cols kk+16..kk+31 (contiguous k order, same for A and B; B fed as N-major rows)
// ---------------------------------------------------------------------------
__device__ __forceinline__ v16bf load_frag(const __bf16* __restrict__ base,
                                           int row_stride, int lane, int kk) {
  const __bf16* p = base + (size_t)(lane & 15) * row_stride + kk + ((lane >> 4) << 4);
  return *(const v16bf*)p;
}
__device__ __forceinline__ v16bf lds_frag(const __bf16* base, int row_stride,
                                          int lane, int kk) {
  const __bf16* p = base + (lane & 15) * row_stride + kk + ((lane >> 4) << 4);
  return *(const v16bf*)p;   // ds_load_b128 x2
}

__device__ __forceinline__ float sigf(float x) { return 1.0f / (1.0f + __expf(-x)); }

// ---------------------------------------------------------------------------
// Grid-wide barrier for the persistent LSTM kernel
// ---------------------------------------------------------------------------
__device__ __forceinline__ void grid_barrier(unsigned* counter, unsigned* gen,
                                             unsigned nblocks) {
  __syncthreads();
  if (threadIdx.x == 0) {
    __threadfence();
    unsigned g = __hip_atomic_load(gen, __ATOMIC_ACQUIRE, __HIP_MEMORY_SCOPE_AGENT);
    unsigned a = __hip_atomic_fetch_add(counter, 1u, __ATOMIC_ACQ_REL,
                                        __HIP_MEMORY_SCOPE_AGENT);
    if (a == nblocks - 1u) {
      __hip_atomic_store(counter, 0u, __ATOMIC_RELAXED, __HIP_MEMORY_SCOPE_AGENT);
      __hip_atomic_fetch_add(gen, 1u, __ATOMIC_RELEASE, __HIP_MEMORY_SCOPE_AGENT);
    } else {
      while (__hip_atomic_load(gen, __ATOMIC_ACQUIRE, __HIP_MEMORY_SCOPE_AGENT) == g) {
        __builtin_amdgcn_s_sleep(2);
      }
    }
  }
  __syncthreads();
}

// ---------------------------------------------------------------------------
// K1: f32 -> bf16 weight conversion
// ---------------------------------------------------------------------------
__global__ void cvt_bf16(const float* __restrict__ src, __bf16* __restrict__ dst,
                         size_t n) {
  size_t stride = (size_t)gridDim.x * blockDim.x;
  for (size_t i = (size_t)blockIdx.x * blockDim.x + threadIdx.x; i < n; i += stride)
    dst[i] = (__bf16)src[i];
}

// ---------------------------------------------------------------------------
// K2: embedding gather -> bf16
// ---------------------------------------------------------------------------
__global__ void gather_embed(const int* __restrict__ x,
                             const float* __restrict__ embW,
                             __bf16* __restrict__ emb) {
  const size_t n = (size_t)BB * SS * EE;
  size_t stride = (size_t)gridDim.x * blockDim.x;
  for (size_t i = (size_t)blockIdx.x * blockDim.x + threadIdx.x; i < n; i += stride) {
    size_t tok = i >> 9;            // / EE
    int    e   = (int)(i & (EE - 1));
    emb[i] = (__bf16)embW[(size_t)x[tok] * EE + e];
  }
}

// ---------------------------------------------------------------------------
// K3: input projection GEMM: gates[32768, 2048] = emb @ Wi^T + bias
// All 8 waves in a block share the same 64-col B strip: stage it in LDS via
// async-to-LDS, double-buffered across K chunks.
// ---------------------------------------------------------------------------
__global__ void input_gemm(const __bf16* __restrict__ emb,
                           const __bf16* __restrict__ Wi,   // [2048, 512]
                           const float*  __restrict__ bias, // [2048]
                           float* __restrict__ gates) {
  __shared__ __bf16 ldsB[2][64 * 32];   // 2 x 4KB: 64 gate cols x 32 K values
  int tid = threadIdx.x;
  int wave = tid >> 5, lane = tid & 31;
  int mtile = blockIdx.x * 8 + wave;
  int n0    = blockIdx.y * 64;
  const __bf16* Abase = emb + (size_t)mtile * 16 * EE;

  // stage(buf, kk): 256 threads x 16B = 4KB strip
  auto stage = [&](int buf, int kk) {
    int row = tid >> 2, c4 = tid & 3;   // 64 rows x 4 chunks
    async_g2l_b128(lds_addr(&ldsB[buf][row * 32 + c4 * 8]),
                   (const void*)(Wi + (size_t)(n0 + row) * EE + kk + c4 * 8));
  };

  stage(0, 0);
  wait_async0();
  __syncthreads();

  v8f acc[4] = {};
  int buf = 0;
  for (int kk = 0; kk < EE; kk += 32) {
    if (kk + 32 < EE) stage(buf ^ 1, kk + 32);
    v16bf a = load_frag(Abase, EE, lane, kk);
    if (kk + 32 < EE)
      __builtin_prefetch((const void*)(Abase + (size_t)(lane & 15) * EE + kk + 32), 0, 0);
#pragma unroll
    for (int j = 0; j < 4; ++j) {
      v16bf b = lds_frag(&ldsB[buf][j * 16 * 32], 32, lane, 0);
      acc[j] = __builtin_amdgcn_wmma_f32_16x16x32_bf16(false, a, false, b,
                                                       (short)0, acc[j], false, false);
    }
    if (kk + 32 < EE) { wait_async0(); __syncthreads(); buf ^= 1; }
  }

  int col = lane & 15, rhi = (lane >> 4) ? 8 : 0;
#pragma unroll
  for (int j = 0; j < 4; ++j) {
    int n = n0 + j * 16 + col;
    float bv = bias[n];
#pragma unroll
    for (int r = 0; r < 8; ++r) {
      int m = mtile * 16 + r + rhi;
      gates[(size_t)m * N4H + n] = acc[j][r] + bv;
    }
  }
}

// ---------------------------------------------------------------------------
// K4: persistent bidirectional LSTM. 16 blocks: 0-7 fwd, 8-15 bwd.
// Per step: h_prev (16x512 bf16 = 16KB) staged once into LDS via async DMA,
// consumed by all 8 waves for all 16 gate tiles (A fragments via ds_load) and
// by the pointwise stage; B = Wh rows from global (L2-resident).
// ---------------------------------------------------------------------------
__global__ void lstm_kernel(const __bf16* __restrict__ WhF,
                            const __bf16* __restrict__ WhB,     // [2048,512]
                            const float*  __restrict__ gatesF,  // [S*B, 2048]
                            const float*  __restrict__ gatesB,
                            const unsigned char* __restrict__ mask, // [B,S]
                            __bf16* __restrict__ hbuf,   // [dir][phase][B][H]
                            __bf16* __restrict__ hcat,   // [B][S][ENC]
                            unsigned* __restrict__ ctrl, int nblocks) {
  int dir = blockIdx.x >> 3;
  int blk = blockIdx.x & 7;
  const __bf16* Wh       = dir ? WhB : WhF;
  const float*  gates_in = dir ? gatesB : gatesF;
  __bf16* hb = hbuf + (size_t)dir * 2 * BB * HH;

  __shared__ __bf16 ldsH[BB * HH];   // 16KB staged h_prev
  __shared__ float  gbuf[16 * 256];  // gate pre-activations
  __shared__ float  cbuf[16 * 64];   // cell state slice
  for (int i = threadIdx.x; i < 16 * 64; i += blockDim.x) cbuf[i] = 0.f;

  int tid = threadIdx.x;
  int wave = tid >> 5, lane = tid & 31;
  int col = lane & 15, rhi = (lane >> 4) ? 8 : 0;

  for (int t = 0; t < SS; ++t) {
    int tt = dir ? (SS - 1 - t) : t;
    const __bf16* hprev = hb + (size_t)(t & 1) * BB * HH;
    __bf16*       hnext = hb + (size_t)((t & 1) ^ 1) * BB * HH;

    // async-stage h_prev: 1024 16B chunks over 256 threads (4 each)
#pragma unroll
    for (int k = 0; k < 4; ++k) {
      int chunk = tid + 256 * k;
      async_g2l_b128(lds_addr(&ldsH[chunk * 8]),
                     (const void*)(hprev + chunk * 8));
    }
    wait_async0();
    __syncthreads();

#pragma unroll
    for (int w2 = 0; w2 < 2; ++w2) {
      int tile = wave * 2 + w2;          // 0..15
      int g    = tile >> 2;              // gate (i,f,g,o)
      int loc0 = (tile & 3) * 16;
      int n0   = g * HH + blk * 64 + loc0;
      v8f acc = {};
      for (int kk = 0; kk < HH; kk += 32) {
        v16bf a = lds_frag(ldsH, HH, lane, kk);
        v16bf b = load_frag(Wh + (size_t)n0 * HH, HH, lane, kk);
        acc = __builtin_amdgcn_wmma_f32_16x16x32_bf16(false, a, false, b,
                                                      (short)0, acc, false, false);
      }
      int lcol = g * 64 + loc0 + col;
#pragma unroll
      for (int r = 0; r < 8; ++r) {
        int m = r + rhi;                 // batch index
        float pre = acc[r] + gates_in[((size_t)tt * BB + m) * N4H + (n0 + col)];
        gbuf[m * 256 + lcol] = pre;
      }
    }
    __syncthreads();

    // pointwise: 16 batches x 64 units
    for (int p = tid; p < 16 * 64; p += blockDim.x) {
      int b = p >> 6, jj = p & 63;
      int j = blk * 64 + jj;
      float ig = sigf(gbuf[b * 256 + jj]);
      float fg = sigf(gbuf[b * 256 + 64 + jj]);
      float gg = tanhf(gbuf[b * 256 + 128 + jj]);
      float og = sigf(gbuf[b * 256 + 192 + jj]);
      float c  = cbuf[p];
      float c2 = fg * c + ig * gg;
      float h2 = og * tanhf(c2);
      float mt = mask[(size_t)b * SS + tt] ? 1.f : 0.f;
      float hp = (float)ldsH[b * HH + j];
      cbuf[p] = mt * c2 + (1.f - mt) * c;
      hnext[(size_t)b * HH + j] = (__bf16)(mt * h2 + (1.f - mt) * hp);
      hcat[((size_t)b * SS + tt) * ENC + (size_t)dir * HH + j] = (__bf16)(mt * h2);
    }
    grid_barrier(ctrl, ctrl + 1, (unsigned)nblocks);
  }
}

// ---------------------------------------------------------------------------
// K5: fused scorer: score[bs] = (1/32)*sum_e tanh((h@W_s^T)[bs,e]+b_s[e])*v_s[e]
// One block per 16-row tile; A tile (16x1024 = 32KB) staged in LDS via async
// DMA, shared by all 8 waves (8x reuse); B strips from global (L2-resident).
// ---------------------------------------------------------------------------
__global__ void scorer(const __bf16* __restrict__ hcat,   // [32768, 1024]
                       const __bf16* __restrict__ Ws,     // [1024, 1024]
                       const float*  __restrict__ b_s,
                       const float*  __restrict__ v_s,
                       float* __restrict__ score) {
  __shared__ __bf16 ldsA[16 * ENC];    // 32KB
  __shared__ float  rowsum[16];
  int tid = threadIdx.x;
  if (tid < 16) rowsum[tid] = 0.f;

  int mtile = blockIdx.x;
  const __bf16* Abase = hcat + (size_t)mtile * 16 * ENC;
  // stage A: 2048 16B chunks over 256 threads (8 each)
#pragma unroll
  for (int k = 0; k < 8; ++k) {
    int chunk = tid + 256 * k;
    async_g2l_b128(lds_addr(&ldsA[chunk * 8]), (const void*)(Abase + chunk * 8));
  }
  wait_async0();
  __syncthreads();

  int wave = tid >> 5, lane = tid & 31;
  int col = lane & 15, rhi = (lane >> 4) ? 8 : 0;
  int nbase = wave * 128;

  v8f acc[8] = {};
  for (int kk = 0; kk < ENC; kk += 32) {
    v16bf a = lds_frag(ldsA, ENC, lane, kk);
#pragma unroll
    for (int j = 0; j < 8; ++j) {
      v16bf b = load_frag(Ws + (size_t)(nbase + j * 16) * ENC, ENC, lane, kk);
      acc[j] = __builtin_amdgcn_wmma_f32_16x16x32_bf16(false, a, false, b,
                                                       (short)0, acc[j], false, false);
    }
  }
#pragma unroll
  for (int r = 0; r < 8; ++r) {
    int m = r + rhi;
    float local = 0.f;
#pragma unroll
    for (int j = 0; j < 8; ++j) {
      int n = nbase + j * 16 + col;
      local += tanhf(acc[j][r] + b_s[n]) * v_s[n];
    }
    atomicAdd(&rowsum[m], local);        // ds_add_f32
  }
  __syncthreads();
  if (tid < 16)
    score[(size_t)mtile * 16 + tid] = rowsum[tid] * 0.03125f; // 1/sqrt(1024)
}

// ---------------------------------------------------------------------------
// K6: sparsemax per batch row via bisection on tau: sum(relu(z - tau)) = 1
// ---------------------------------------------------------------------------
__global__ void sparsemax_kernel(const float* __restrict__ score,
                                 const unsigned char* __restrict__ mask,
                                 float* __restrict__ out) {
  __shared__ float z[SS];
  __shared__ float red[256];
  __shared__ float tlo, thi;
  int b = blockIdx.x, tid = threadIdx.x;

  for (int i = tid; i < SS; i += blockDim.x) z[i] = score[(size_t)b * SS + i];
  __syncthreads();

  float m = -3.402823466e38f;
  for (int i = tid; i < SS; i += blockDim.x) m = fmaxf(m, z[i]);
  red[tid] = m; __syncthreads();
  for (int s = 128; s > 0; s >>= 1) {
    if (tid < s) red[tid] = fmaxf(red[tid], red[tid + s]);
    __syncthreads();
  }
  if (tid == 0) { tlo = red[0] - 1.f; thi = red[0]; }
  __syncthreads();

  for (int it = 0; it < 44; ++it) {
    float tau = 0.5f * (tlo + thi);
    float s = 0.f;
    for (int i = tid; i < SS; i += blockDim.x) s += fmaxf(z[i] - tau, 0.f);
    red[tid] = s; __syncthreads();
    for (int st = 128; st > 0; st >>= 1) {
      if (tid < st) red[tid] += red[tid + st];
      __syncthreads();
    }
    if (tid == 0) { if (red[0] > 1.f) tlo = tau; else thi = tau; }
    __syncthreads();
  }
  float tau = 0.5f * (tlo + thi);
  for (int i = tid; i < SS; i += blockDim.x) {
    float v = fmaxf(z[i] - tau, 0.f);
    out[(size_t)b * SS + i] = mask[(size_t)b * SS + i] ? v : 0.f;
  }
}

// ---------------------------------------------------------------------------
// Host launcher
// ---------------------------------------------------------------------------
extern "C" void kernel_launch(void* const* d_in, const int* in_sizes, int n_in,
                              void* d_out, int out_size, void* d_ws, size_t ws_size,
                              hipStream_t stream) {
  (void)in_sizes; (void)n_in; (void)out_size; (void)ws_size;

  const int*           x    = (const int*)d_in[0];
  const unsigned char* mask = (const unsigned char*)d_in[1];
  const float* embW = (const float*)d_in[2];
  const float* Wi_f = (const float*)d_in[3];
  const float* Wh_f = (const float*)d_in[4];
  const float* b_f  = (const float*)d_in[5];
  const float* Wi_b = (const float*)d_in[6];
  const float* Wh_b = (const float*)d_in[7];
  const float* b_b  = (const float*)d_in[8];
  const float* W_s  = (const float*)d_in[9];
  const float* b_s  = (const float*)d_in[10];
  const float* v_s  = (const float*)d_in[11];

  char* p = (char*)d_ws;
  auto carve = [&](size_t bytes) -> void* {
    void* r = (void*)p;
    p += (bytes + 255) & ~(size_t)255;
    return r;
  };
  unsigned* ctrl   = (unsigned*)carve(256);
  __bf16*   hbuf   = (__bf16*)carve((size_t)2 * 2 * BB * HH * sizeof(__bf16));
  size_t    zbytes = (size_t)(p - (char*)ctrl);    // barrier ctrl + h(0)=0
  __bf16*   emb    = (__bf16*)carve((size_t)BB * SS * EE * sizeof(__bf16));
  __bf16*   WiFb   = (__bf16*)carve((size_t)N4H * EE * sizeof(__bf16));
  __bf16*   WiBb   = (__bf16*)carve((size_t)N4H * EE * sizeof(__bf16));
  __bf16*   WhFb   = (__bf16*)carve((size_t)N4H * HH * sizeof(__bf16));
  __bf16*   WhBb   = (__bf16*)carve((size_t)N4H * HH * sizeof(__bf16));
  __bf16*   Wsb    = (__bf16*)carve((size_t)ENC * ENC * sizeof(__bf16));
  float*    gatesF = (float*)carve((size_t)BB * SS * N4H * sizeof(float));
  float*    gatesB = (float*)carve((size_t)BB * SS * N4H * sizeof(float));
  __bf16*   hcat   = (__bf16*)carve((size_t)BB * SS * ENC * sizeof(__bf16));
  float*    scoreb = (float*)carve((size_t)BB * SS * sizeof(float));

  hipMemsetAsync(ctrl, 0, zbytes, stream);

  cvt_bf16<<<512, 256, 0, stream>>>(Wi_f, WiFb, (size_t)N4H * EE);
  cvt_bf16<<<512, 256, 0, stream>>>(Wi_b, WiBb, (size_t)N4H * EE);
  cvt_bf16<<<512, 256, 0, stream>>>(Wh_f, WhFb, (size_t)N4H * HH);
  cvt_bf16<<<512, 256, 0, stream>>>(Wh_b, WhBb, (size_t)N4H * HH);
  cvt_bf16<<<512, 256, 0, stream>>>(W_s,  Wsb,  (size_t)ENC * ENC);

  gather_embed<<<4096, 256, 0, stream>>>(x, embW, emb);

  input_gemm<<<dim3(256, 32), 256, 0, stream>>>(emb, WiFb, b_f, gatesF);
  input_gemm<<<dim3(256, 32), 256, 0, stream>>>(emb, WiBb, b_b, gatesB);

  lstm_kernel<<<16, 256, 0, stream>>>(WhFb, WhBb, gatesF, gatesB, mask,
                                      hbuf, hcat, ctrl, 16);

  scorer<<<BB * SS / 16, 256, 0, stream>>>(hcat, Wsb, b_s, v_s, scoreb);

  sparsemax_kernel<<<BB, 256, 0, stream>>>(scoreb, mask, (float*)d_out);
}
